// VssblockdownModule_32521492365671
// MI455X (gfx1250) — matrix operand order repaired and checked by reading, not compiled, not run on hardware
//
#include <hip/hip_runtime.h>
#include <hip/hip_bf16.h>

// ---------------------------------------------------------------------------
// VSS block for MI455X (gfx1250, wave32, WMMA + TDM).
// GEMMs: v_wmma_f32_16x16x32_f16, b128 fragment loads, pre-swizzled B weights.
// Scan: 3-pass chunked linear recurrence; per-chunk x_dbl slab (256x48 f32,
//       48KB) is DMA'd into LDS via tensor_load_to_lds (TENSORcnt) so the
//       serial recurrence reads rows from LDS instead of global memory.
// ---------------------------------------------------------------------------

typedef __attribute__((ext_vector_type(16))) _Float16 v16h;
typedef __attribute__((ext_vector_type(8)))  _Float16 v8h;
typedef __attribute__((ext_vector_type(8)))  float    v8f;
typedef __attribute__((ext_vector_type(4)))  unsigned int u32x4;
typedef __attribute__((ext_vector_type(8)))  int      i32x8;
typedef __attribute__((ext_vector_type(4)))  int      i32x4;

#define BB   8
#define CC   96
#define DD   192
#define KK   4
#define NN   16
#define RR   6
#define HH   64
#define WW   64
#define LL   4096
#define CPAD 48           // R+2N = 38 padded to 48
#define NCH  16           // scan chunks
#define CHL  256          // chunk length

__device__ __forceinline__ float wave_sum(float v) {
#pragma unroll
  for (int off = 16; off > 0; off >>= 1) v += __shfl_xor(v, off, 32);
  return v;
}

// ---- TDM: 2D f32 tile (rows x cols, row stride == cols) global -> LDS ------
// D# per ISA 8.3/8.4: group0 {count=1, lds_addr, global_addr, type=2};
// group1 {data_size=4B, tensor_dim0/1, tile_dim0/1, dim0_stride}; groups2/3=0.
// clang-23 toolchain: 6-arg builtin (g0, g1, g2, g3, extra, cpol).
__device__ __forceinline__ void tdm_load_2d_f32(const float* gsrc,
                                                unsigned lds_byte_off,
                                                int rows, int cols) {
  unsigned long long ga = (unsigned long long)(uintptr_t)gsrc;
  u32x4 g0;
  g0[0] = 1u;                                        // count=1, user descriptor
  g0[1] = lds_byte_off;                              // LDS address (bytes)
  g0[2] = (unsigned)(ga & 0xFFFFFFFFu);              // global_addr[31:0]
  g0[3] = (unsigned)((ga >> 32) & 0x01FFFFFFu) | (2u << 30);  // addr[56:32]|type=2
  i32x8 g1;
  g1[0] = (2 << 16);                                 // data_size: 2 -> 4 bytes
  g1[1] = (int)((unsigned)(cols & 0xFFFF) << 16);    // tensor_dim0[15:0]
  g1[2] = (int)(((unsigned)cols >> 16) |
                ((unsigned)(rows & 0xFFFF) << 16));  // dim0[31:16] | dim1[15:0]
  g1[3] = (int)(((unsigned)rows >> 16) |
                ((unsigned)(cols & 0xFFFF) << 16));  // dim1[31:16] | tile_dim0
  g1[4] = rows & 0xFFFF;                             // tile_dim1 (tile_dim2=0)
  g1[5] = cols;                                      // tensor_dim0_stride[31:0]
  g1[6] = 0;                                         // stride[47:32] | dim1_stride lo
  g1[7] = 0;
  i32x4 gz4 = {0, 0, 0, 0};
  i32x8 gz8 = {0, 0, 0, 0, 0, 0, 0, 0};
  __builtin_amdgcn_tensor_load_to_lds(g0, g1, gz4, gz4, gz8, 0);
}

// ---- WMMA fragment helpers (wave32) ---------------------------------------
__device__ __forceinline__ v16h load_a_frag(const _Float16* A, int lda, int lane) {
  int m = lane & 15, g = lane >> 4;
  const _Float16* p = A + m * lda + (g << 4);
  v8h lo = *(const v8h*)p;
  v8h hi = *(const v8h*)(p + 8);
  return __builtin_shufflevector(lo, hi, 0, 1, 2, 3, 4, 5, 6, 7,
                                 8, 9, 10, 11, 12, 13, 14, 15);
}
__device__ __forceinline__ v16h load_b_packed(const _Float16* Bp, int frag, int lane) {
  const _Float16* p = Bp + (((size_t)frag << 5) + lane) * 16;
  v8h lo = *(const v8h*)p;
  v8h hi = *(const v8h*)(p + 8);
  return __builtin_shufflevector(lo, hi, 0, 1, 2, 3, 4, 5, 6, 7,
                                 8, 9, 10, 11, 12, 13, 14, 15);
}
__device__ __forceinline__ void store_c(float* C, int ldc, int lane, v8f acc) {
  int n = lane & 15, g = lane >> 4;
#pragma unroll
  for (int v = 0; v < 8; ++v) C[(v + (g << 3)) * ldc + n] = acc[v];
}

template <int KSTEPS, int NT>
__device__ __forceinline__ void gemm_row(const _Float16* Ab, int lda,
                                         const _Float16* Bp, int nt0,
                                         int lane, v8f* acc) {
#pragma unroll
  for (int ks = 0; ks < KSTEPS; ++ks) {
    v16h a = load_a_frag(Ab + ks * 32, lda, lane);
#pragma unroll
    for (int t = 0; t < NT; ++t) {
      v16h b = load_b_packed(Bp, (nt0 + t) * KSTEPS + ks, lane);
      acc[t] = __builtin_amdgcn_wmma_f32_16x16x32_f16(false, a, false, b,
                                                      (short)0, acc[t],
                                                      false, false);
    }
  }
}

// ---------------------------------------------------------------------------
// 0) Weights -> f16, pre-swizzled into B-fragment order.
// ---------------------------------------------------------------------------
__global__ void prep_weights_kernel(const float* in_proj_w, const float* x_proj_w,
                                    const float* out_proj_w, const float* pm_red_w,
                                    _Float16* w_in_t, _Float16* xprojT,
                                    _Float16* outprojT, _Float16* pmT) {
  int t = blockIdx.x * 256 + threadIdx.x;
  if (t < 36864) {
    int u = t;
    int e = u & 15, lane = (u >> 4) & 31, rest = u >> 9;
    int ks = rest % 3;
    int k = 32 * ks + ((lane >> 4) << 4) + e;
    int n = ((rest / 3) << 4) + (lane & 15);
    w_in_t[u] = (_Float16)in_proj_w[n * CC + k];
  } else if (t < 73728) {
    int u = t - 36864;
    int kk = u / 9216, v = u % 9216;
    int e = v & 15, lane = (v >> 4) & 31, rest = v >> 9;
    int ks = rest % 6;
    int k = 32 * ks + ((lane >> 4) << 4) + e;
    int c = ((rest / 6) << 4) + (lane & 15);
    xprojT[u] = (c < RR + 2 * NN)
                  ? (_Float16)x_proj_w[((size_t)kk * (RR + 2 * NN) + c) * DD + k]
                  : (_Float16)0.f;
  } else if (t < 92160) {
    int u = t - 73728;
    int e = u & 15, lane = (u >> 4) & 31, rest = u >> 9;
    int ks = rest % 6;
    int k = 32 * ks + ((lane >> 4) << 4) + e;
    int n = ((rest / 6) << 4) + (lane & 15);
    outprojT[u] = (_Float16)out_proj_w[n * DD + k];
  } else if (t < 165888) {
    int u = t - 92160;
    int e = u & 15, lane = (u >> 4) & 31, rest = u >> 9;
    int ks = rest % 12;
    int k = 32 * ks + ((lane >> 4) << 4) + e;
    int n = ((rest / 12) << 4) + (lane & 15);
    pmT[u] = (_Float16)pm_red_w[n * 384 + k];
  }
}

// ---------------------------------------------------------------------------
// 1) LayerNorm over C of transposed x -> xn (f16).
// ---------------------------------------------------------------------------
__global__ void ln_in_kernel(const float* x, const float* w, const float* bia,
                             _Float16* xn_h) {
  int wave = threadIdx.x >> 5, lane = threadIdx.x & 31;
  int row = blockIdx.x * 8 + wave;
  int b = row >> 12, l = row & 4095;
  const float* xb = x + (size_t)b * CC * LL + l;
  float v0 = xb[(size_t)lane * LL];
  float v1 = xb[(size_t)(lane + 32) * LL];
  float v2 = xb[(size_t)(lane + 64) * LL];
  float mu = wave_sum(v0 + v1 + v2) * (1.f / 96.f);
  float d0 = v0 - mu, d1 = v1 - mu, d2 = v2 - mu;
  float var = wave_sum(d0 * d0 + d1 * d1 + d2 * d2) * (1.f / 96.f);
  float rs = rsqrtf(var + 1e-5f);
  _Float16* o = xn_h + (size_t)row * CC;
  o[lane]      = (_Float16)(d0 * rs * w[lane]      + bia[lane]);
  o[lane + 32] = (_Float16)(d1 * rs * w[lane + 32] + bia[lane + 32]);
  o[lane + 64] = (_Float16)(d2 * rs * w[lane + 64] + bia[lane + 64]);
}

// ---------------------------------------------------------------------------
// 2) xz = xn @ in_proj_w.T : (32768x96)x(96x384). NT=3.
// ---------------------------------------------------------------------------
__global__ void gemm_xz_kernel(const _Float16* A, const _Float16* Bp, float* C) {
  int lane = threadIdx.x & 31;
  int id = blockIdx.x * 8 + (threadIdx.x >> 5);
  int mt = id >> 3, nt0 = (id & 7) * 3;
  v8f acc[3] = {};
  gemm_row<3, 3>(A + (size_t)mt * 16 * CC, CC, Bp, nt0, lane, acc);
#pragma unroll
  for (int t = 0; t < 3; ++t)
    store_c(C + (size_t)mt * 16 * 384 + (nt0 + t) * 16, 384, lane, acc[t]);
}

// ---------------------------------------------------------------------------
// 3) Depthwise 3x3 conv + bias + SiLU -> 4 direction layouts (b,k,l,d) f16.
// ---------------------------------------------------------------------------
__global__ void conv_silu_kernel(const float* xz, const float* cw, const float* cb,
                                 _Float16* xs_h) {
  int d = threadIdx.x;
  int w = blockIdx.x, h = blockIdx.y, b = blockIdx.z;
  float acc = cb[d];
  const float* wt = cw + d * 9;
#pragma unroll
  for (int dh = -1; dh <= 1; ++dh)
#pragma unroll
    for (int dw = -1; dw <= 1; ++dw) {
      int hh = h + dh, ww = w + dw;
      if (hh >= 0 && hh < HH && ww >= 0 && ww < WW) {
        float v = xz[((size_t)b * LL + hh * WW + ww) * 384 + d];
        acc += v * wt[(dh + 1) * 3 + (dw + 1)];
      }
    }
  float sv = acc / (1.f + __expf(-acc));
  _Float16 hv = (_Float16)sv;
  int l0 = h * WW + w, l1 = w * HH + h;
  size_t base = (size_t)b * KK * LL * DD;
  xs_h[base + ((size_t)0 * LL + l0) * DD + d] = hv;
  xs_h[base + ((size_t)1 * LL + l1) * DD + d] = hv;
  xs_h[base + ((size_t)2 * LL + (LL - 1 - l0)) * DD + d] = hv;
  xs_h[base + ((size_t)3 * LL + (LL - 1 - l1)) * DD + d] = hv;
}

// ---------------------------------------------------------------------------
// 4) x_dbl per (b,k): (4096x192)x(192x48). NT=3.
// ---------------------------------------------------------------------------
__global__ void gemm_xdbl_kernel(const _Float16* xs_h, const _Float16* xprojT,
                                 float* x_dbl) {
  int lane = threadIdx.x & 31;
  int bk = blockIdx.y;
  int mt = blockIdx.x * 8 + (threadIdx.x >> 5);
  const _Float16* A = xs_h + (size_t)bk * LL * DD + (size_t)mt * 16 * DD;
  const _Float16* Bp = xprojT + (bk & 3) * (DD * CPAD);
  v8f acc[3] = {};
  gemm_row<6, 3>(A, DD, Bp, 0, lane, acc);
  float* Cb = x_dbl + (size_t)bk * LL * CPAD + (size_t)mt * 16 * CPAD;
#pragma unroll
  for (int t = 0; t < 3; ++t) store_c(Cb + t * 16, CPAD, lane, acc[t]);
}

// ---------------------------------------------------------------------------
// 5a) Scan pass 1: TDM the chunk's x_dbl slab into LDS, then local scan.
// ---------------------------------------------------------------------------
__global__ void scan_chunk_kernel(const float* x_dbl, const _Float16* xs_h,
                                  const float* dtw_all, const float* dtb_all,
                                  const float* A_logs, const float* Ds,
                                  float* y_buf, float* hloc, float* sdel) {
  __shared__ float srow[CHL * CPAD];          // 48 KB
  int d = threadIdx.x;
  int ch = blockIdx.x, bk = blockIdx.y, k = bk & 3;
  size_t l0 = (size_t)ch * CHL;

  if ((threadIdx.x >> 5) == 0) {              // wave 0 launches the DMA
    tdm_load_2d_f32(x_dbl + ((size_t)bk * LL + l0) * CPAD, 0, CHL, CPAD);
    __builtin_amdgcn_s_wait_tensorcnt(0);
  }
  float dtw[RR];
#pragma unroll
  for (int r = 0; r < RR; ++r) dtw[r] = dtw_all[(k * DD + d) * RR + r];
  float dtb = dtb_all[k * DD + d];
  float Aw[NN];
#pragma unroll
  for (int n = 0; n < NN; ++n) Aw[n] = -__expf(A_logs[((size_t)k * DD + d) * NN + n]);
  float Dv = Ds[k * DD + d];
  float hst[NN];
#pragma unroll
  for (int n = 0; n < NN; ++n) hst[n] = 0.f;
  __syncthreads();                            // LDS slab visible to all waves

  const _Float16* u_p = xs_h + ((size_t)bk * LL + l0) * DD + d;
  float*          y_p = y_buf + ((size_t)bk * LL + l0) * DD + d;
  float sumd = 0.f;

  for (int l = 0; l < CHL; ++l) {
    const float* row = srow + l * CPAD;       // LDS, wave-uniform broadcast
    float dt = dtb;
#pragma unroll
    for (int r = 0; r < RR; ++r) dt += row[r] * dtw[r];
    float delta = (dt > 20.f) ? dt : log1pf(__expf(dt));
    sumd += delta;
    float u  = (float)u_p[(size_t)l * DD];
    float du = delta * u;
    float acc = 0.f;
#pragma unroll
    for (int n = 0; n < NN; ++n) {
      hst[n] = hst[n] * __expf(delta * Aw[n]) + du * row[RR + n];
      acc += hst[n] * row[RR + NN + n];
    }
    y_p[(size_t)l * DD] = acc + Dv * u;
  }
  size_t base = ((size_t)(bk * NCH + ch) * DD + d) * NN;
#pragma unroll
  for (int n = 0; n < NN; ++n) hloc[base + n] = hst[n];
  sdel[(bk * NCH + ch) * DD + d] = sumd;
}

// ---------------------------------------------------------------------------
// 5b) Combine chunk states (16 sequential steps).
// ---------------------------------------------------------------------------
__global__ void scan_combine_kernel(const float* hloc, const float* sdel,
                                    const float* A_logs, float* h0buf) {
  int d = threadIdx.x, bk = blockIdx.x, k = bk & 3;
  float Aw[NN];
#pragma unroll
  for (int n = 0; n < NN; ++n) Aw[n] = -__expf(A_logs[((size_t)k * DD + d) * NN + n]);
  float h0[NN];
#pragma unroll
  for (int n = 0; n < NN; ++n) h0[n] = 0.f;
  for (int ch = 0; ch < NCH; ++ch) {
    size_t base = ((size_t)(bk * NCH + ch) * DD + d) * NN;
    float sd = sdel[(bk * NCH + ch) * DD + d];
#pragma unroll
    for (int n = 0; n < NN; ++n) {
      h0buf[base + n] = h0[n];
      h0[n] = h0[n] * __expf(Aw[n] * sd) + hloc[base + n];
    }
  }
}

// ---------------------------------------------------------------------------
// 5c) Fix-up: y += C . (h0 * exp(A * cum_delta)); TDM slab into LDS again.
// ---------------------------------------------------------------------------
__global__ void scan_fix_kernel(const float* x_dbl, const float* dtw_all,
                                const float* dtb_all, const float* A_logs,
                                const float* h0buf, float* y_buf) {
  __shared__ float srow[CHL * CPAD];
  int d = threadIdx.x;
  int ch = blockIdx.x + 1, bk = blockIdx.y, k = bk & 3;
  size_t l0 = (size_t)ch * CHL;

  if ((threadIdx.x >> 5) == 0) {
    tdm_load_2d_f32(x_dbl + ((size_t)bk * LL + l0) * CPAD, 0, CHL, CPAD);
    __builtin_amdgcn_s_wait_tensorcnt(0);
  }
  float dtw[RR];
#pragma unroll
  for (int r = 0; r < RR; ++r) dtw[r] = dtw_all[(k * DD + d) * RR + r];
  float dtb = dtb_all[k * DD + d];
  float Aw[NN];
#pragma unroll
  for (int n = 0; n < NN; ++n) Aw[n] = -__expf(A_logs[((size_t)k * DD + d) * NN + n]);
  float h0[NN];
  size_t hb = ((size_t)(bk * NCH + ch) * DD + d) * NN;
#pragma unroll
  for (int n = 0; n < NN; ++n) h0[n] = h0buf[hb + n];
  __syncthreads();

  float* y_p = y_buf + ((size_t)bk * LL + l0) * DD + d;
  float cumd = 0.f;
  for (int l = 0; l < CHL; ++l) {
    const float* row = srow + l * CPAD;
    float dt = dtb;
#pragma unroll
    for (int r = 0; r < RR; ++r) dt += row[r] * dtw[r];
    float delta = (dt > 20.f) ? dt : log1pf(__expf(dt));
    cumd += delta;
    float corr = 0.f;
#pragma unroll
    for (int n = 0; n < NN; ++n)
      corr += row[RR + NN + n] * h0[n] * __expf(Aw[n] * cumd);
    y_p[(size_t)l * DD] += corr;
  }
}

// ---------------------------------------------------------------------------
// 6) Merge 4 directions + out_norm LN + SiLU(z) gate -> yo (f16).
// ---------------------------------------------------------------------------
__global__ void combine_kernel(const float* y_buf, const float* xz,
                               const float* onw, const float* onb, _Float16* yo_h) {
  int wave = threadIdx.x >> 5, lane = threadIdx.x & 31;
  int row = blockIdx.x * 8 + wave;
  int b = row >> 12, l = row & 4095;
  int l1 = (l & 63) * 64 + (l >> 6);
  size_t base = (size_t)b * KK * LL * DD;
  const float* p0 = y_buf + base + (size_t)l * DD;
  const float* p1 = y_buf + base + ((size_t)LL + l1) * DD;
  const float* p2 = y_buf + base + ((size_t)2 * LL + (LL - 1 - l)) * DD;
  const float* p3 = y_buf + base + ((size_t)3 * LL + (LL - 1 - l1)) * DD;
  float v[6];
  float s = 0.f;
#pragma unroll
  for (int j = 0; j < 6; ++j) {
    int d = lane + 32 * j;
    v[j] = p0[d] + p1[d] + p2[d] + p3[d];
    s += v[j];
  }
  float mu = wave_sum(s) * (1.f / 192.f);
  float vs = 0.f;
#pragma unroll
  for (int j = 0; j < 6; ++j) { v[j] -= mu; vs += v[j] * v[j]; }
  float rs = rsqrtf(wave_sum(vs) * (1.f / 192.f) + 1e-5f);
#pragma unroll
  for (int j = 0; j < 6; ++j) {
    int d = lane + 32 * j;
    float z = xz[(size_t)row * 384 + DD + d];
    float g = z / (1.f + __expf(-z));
    yo_h[(size_t)row * DD + d] = (_Float16)((v[j] * rs * onw[d] + onb[d]) * g);
  }
}

// ---------------------------------------------------------------------------
// 7) out = resid + yo @ out_proj_w.T : (32768x192)x(192x96). NT=6.
// ---------------------------------------------------------------------------
__global__ void gemm_out_kernel(const _Float16* yo_h, const _Float16* opT,
                                const float* x, float* outb) {
  int lane = threadIdx.x & 31;
  int mt = blockIdx.x * 8 + (threadIdx.x >> 5);
  v8f acc[6] = {};
  gemm_row<6, 6>(yo_h + (size_t)mt * 16 * DD, DD, opT, 0, lane, acc);
  int g = lane >> 4;
#pragma unroll
  for (int t = 0; t < 6; ++t) {
    int n = t * 16 + (lane & 15);
#pragma unroll
    for (int v = 0; v < 8; ++v) {
      int m = mt * 16 + v + (g << 3);
      int b = m >> 12, l = m & 4095;
      float r = x[((size_t)b * CC + n) * LL + l];
      outb[(size_t)m * CC + n] = acc[t][v] + r;
    }
  }
}

// ---------------------------------------------------------------------------
// 8) Patch-merge gather + LN -> xm (f16).
// ---------------------------------------------------------------------------
__global__ void pm_prep_kernel(const float* outb, const float* pw, const float* pb,
                               _Float16* xm_h) {
  int wave = threadIdx.x >> 5, lane = threadIdx.x & 31;
  int row = blockIdx.x * 8 + wave;
  int b = row >> 10, ij = row & 1023;
  int i = ij >> 5, j = ij & 31;
  float v[12];
  float s = 0.f;
#pragma unroll
  for (int t = 0; t < 12; ++t) {
    int c = lane + 32 * t;
    int q = c / 96, cc = c % 96;
    int h = 2 * i + (q & 1), w = 2 * j + (q >> 1);
    v[t] = outb[((size_t)b * LL + h * WW + w) * CC + cc];
    s += v[t];
  }
  float mu = wave_sum(s) * (1.f / 384.f);
  float vs = 0.f;
#pragma unroll
  for (int t = 0; t < 12; ++t) { v[t] -= mu; vs += v[t] * v[t]; }
  float rs = rsqrtf(wave_sum(vs) * (1.f / 384.f) + 1e-5f);
#pragma unroll
  for (int t = 0; t < 12; ++t) {
    int c = lane + 32 * t;
    xm_h[(size_t)row * 384 + c] = (_Float16)(v[t] * rs * pw[c] + pb[c]);
  }
}

// ---------------------------------------------------------------------------
// 9) out = xm @ pm_red_w.T, transposed store to (B, 2C, 32, 32). NT=4.
// ---------------------------------------------------------------------------
__global__ void gemm_pm_kernel(const _Float16* xm_h, const _Float16* pmT,
                               float* dout) {
  int lane = threadIdx.x & 31;
  int tile = blockIdx.x * 8 + (threadIdx.x >> 5);
  int mt = tile / 3, nt0 = (tile % 3) * 4;
  v8f acc[4] = {};
  gemm_row<12, 4>(xm_h + (size_t)mt * 16 * 384, 384, pmT, nt0, lane, acc);
  int g = lane >> 4;
#pragma unroll
  for (int t = 0; t < 4; ++t) {
    int n = (nt0 + t) * 16 + (lane & 15);
#pragma unroll
    for (int v = 0; v < 8; ++v) {
      int m = mt * 16 + v + (g << 3);
      int b = m >> 10, ij = m & 1023;
      dout[((size_t)b * DD + n) * 1024 + ij] = acc[t][v];
    }
  }
}

// ---------------------------------------------------------------------------
extern "C" void kernel_launch(void* const* d_in, const int* in_sizes, int n_in,
                              void* d_out, int out_size, void* d_ws, size_t ws_size,
                              hipStream_t stream) {
  const float* x         = (const float*)d_in[0];
  const float* ln1_w     = (const float*)d_in[1];
  const float* ln1_b     = (const float*)d_in[2];
  const float* in_proj_w = (const float*)d_in[3];
  const float* conv_w    = (const float*)d_in[4];
  const float* conv_b    = (const float*)d_in[5];
  const float* x_proj_w  = (const float*)d_in[6];
  const float* dt_pw     = (const float*)d_in[7];
  const float* dt_pb     = (const float*)d_in[8];
  const float* A_logs    = (const float*)d_in[9];
  const float* Ds        = (const float*)d_in[10];
  const float* onw       = (const float*)d_in[11];
  const float* onb       = (const float*)d_in[12];
  const float* out_pw    = (const float*)d_in[13];
  const float* pm_nw     = (const float*)d_in[14];
  const float* pm_nb     = (const float*)d_in[15];
  const float* pm_rw     = (const float*)d_in[16];
  float* out = (float*)d_out;

  char* ws = (char*)d_ws;
  size_t off = 0;
  auto carve = [&](size_t bytes) -> char* {
    char* p = ws + off;
    off += (bytes + 255) & ~(size_t)255;
    return p;
  };
  _Float16* xn_h     = (_Float16*)carve((size_t)32768 * CC * 2);
  _Float16* w_in_t   = (_Float16*)carve((size_t)CC * 384 * 2);
  _Float16* xprojT   = (_Float16*)carve((size_t)KK * DD * CPAD * 2);
  _Float16* outprojT = (_Float16*)carve((size_t)DD * CC * 2);
  _Float16* pmT      = (_Float16*)carve((size_t)384 * DD * 2);
  float*    xz       = (float*)   carve((size_t)32768 * 384 * 4);
  _Float16* xs_h     = (_Float16*)carve((size_t)BB * KK * LL * DD * 2);
  float*    x_dbl    = (float*)   carve((size_t)BB * KK * LL * CPAD * 4);
  float*    y_buf    = (float*)   carve((size_t)BB * KK * LL * DD * 4);
  float*    hloc     = (float*)   carve((size_t)32 * NCH * DD * NN * 4);
  float*    sdel     = (float*)   carve((size_t)32 * NCH * DD * 4);
  float*    h0buf    = (float*)   carve((size_t)32 * NCH * DD * NN * 4);
  _Float16* yo_h     = (_Float16*)carve((size_t)32768 * DD * 2);
  float*    outb     = (float*)   carve((size_t)32768 * CC * 4);
  _Float16* xm_h     = (_Float16*)carve((size_t)8192 * 384 * 2);

  prep_weights_kernel<<<(165888 + 255) / 256, 256, 0, stream>>>(
      in_proj_w, x_proj_w, out_pw, pm_rw, w_in_t, xprojT, outprojT, pmT);

  ln_in_kernel<<<4096, 256, 0, stream>>>(x, ln1_w, ln1_b, xn_h);

  gemm_xz_kernel<<<2048, 256, 0, stream>>>(xn_h, w_in_t, xz);

  conv_silu_kernel<<<dim3(WW, HH, BB), DD, 0, stream>>>(xz, conv_w, conv_b, xs_h);

  gemm_xdbl_kernel<<<dim3(32, 32), 256, 0, stream>>>(xs_h, xprojT, x_dbl);

  scan_chunk_kernel<<<dim3(NCH, 32), DD, 0, stream>>>(
      x_dbl, xs_h, dt_pw, dt_pb, A_logs, Ds, y_buf, hloc, sdel);

  scan_combine_kernel<<<32, DD, 0, stream>>>(hloc, sdel, A_logs, h0buf);

  scan_fix_kernel<<<dim3(NCH - 1, 32), DD, 0, stream>>>(
      x_dbl, dt_pw, dt_pb, A_logs, h0buf, y_buf);

  combine_kernel<<<4096, 256, 0, stream>>>(y_buf, xz, onw, onb, yo_h);

  gemm_out_kernel<<<256, 256, 0, stream>>>(yo_h, outprojT, x, outb);

  pm_prep_kernel<<<1024, 256, 0, stream>>>(outb, pm_nw, pm_nb, xm_h);

  gemm_pm_kernel<<<192, 256, 0, stream>>>(xm_h, pmT, out);

  (void)in_sizes; (void)n_in; (void)out_size; (void)ws_size;
}
